// TransformerBlock_38628935860379
// MI455X (gfx1250) — matrix operand order repaired and checked by reading, not compiled, not run on hardware
//
#include <hip/hip_runtime.h>
#include <hip/hip_bf16.h>
#include <math.h>

typedef __bf16 bf16;
typedef __attribute__((ext_vector_type(16))) __bf16 v16bf;
typedef __attribute__((ext_vector_type(8)))  float  v8f;
typedef __attribute__((ext_vector_type(4))) unsigned int u32x4;
typedef __attribute__((ext_vector_type(8))) int i32x8;
typedef __attribute__((ext_vector_type(4))) int i32x4;

// chunk permutation for bf16 A-fragment element order: identity (contiguous k order)
__device__ __forceinline__ int chunk_perm(int c) { return c; }
// permute a column index within its 32-wide K-block into A-fragment order
__device__ __forceinline__ int perm32(int i) {
  return (i & ~31) | (chunk_perm((i >> 3) & 3) << 3) | (i & 7);
}

// ---------------------------------------------------------------------------
// TDM: DMA a 2D tile (64 rows x 64B) from global to LDS. D# per ISA 8.3/8.4.
// data_size=8B, tile_dim0=8 qwords, tile_dim1=64 rows, row stride in qwords.
// 6-arg builtin form: (g0, g1, g2, g3, g4, cpol)
// ---------------------------------------------------------------------------
__device__ __forceinline__ void tdm_load_tile(const void* gaddr, unsigned int lds_addr,
                                              unsigned int stride_qw) {
  unsigned long long ga = (unsigned long long)(uintptr_t)gaddr;
  u32x4 g0;
  g0[0] = 1u;                                        // count=1, user mode
  g0[1] = lds_addr;                                  // LDS byte address
  g0[2] = (unsigned int)ga;                          // global_addr[31:0]
  g0[3] = (unsigned int)((ga >> 32) & 0x1FFFFFFu) | (2u << 30);  // addr[56:32] | type=2
  i32x8 g1;
  g1[0] = (int)(3u << 16);                           // workgroup_mask=0, data_size=8B
  g1[1] = (int)(0x4000u << 16);                      // tensor_dim0 = 0x4000 (low16)
  g1[2] = (int)(0x4000u << 16);                      // tensor_dim0 hi=0, tensor_dim1 lo=0x4000
  g1[3] = (int)(8u << 16);                           // tensor_dim1 hi=0, tile_dim0=8 qwords
  g1[4] = (int)64;                                   // tile_dim1=64 rows, tile_dim2=0
  g1[5] = (int)stride_qw;                            // tensor_dim0_stride[31:0]
  g1[6] = 0;                                         // stride hi, dim1_stride lo
  g1[7] = 0;
  i32x4 z4 = {0, 0, 0, 0};
  i32x8 z8 = {0, 0, 0, 0, 0, 0, 0, 0};
  __builtin_amdgcn_tensor_load_to_lds(g0, g1, z4, z4, z8, 0);
}

// ---------------------------------------------------------------------------
// fp32 [K,N] -> bf16 transposed [N,K]
// ---------------------------------------------------------------------------
__global__ void k_cvt_t_bf16(const float* __restrict__ src, bf16* __restrict__ dst,
                             int K, int N) {
  long long tid = (long long)blockIdx.x * blockDim.x + threadIdx.x;
  if (tid >= (long long)K * N) return;
  int k = (int)(tid / N), n = (int)(tid % N);
  dst[(size_t)n * K + k] = (bf16)src[tid];
}

// ---------------------------------------------------------------------------
// LayerNorm (mean-subtract, no beta) -> bf16, columns in A-fragment order.
// One wave per row.
// ---------------------------------------------------------------------------
__global__ __launch_bounds__(256) void k_layernorm_bf16(
    const float* __restrict__ x, const float* __restrict__ gamma,
    bf16* __restrict__ out, int rows, int cols) {
  int wave = threadIdx.x >> 5;
  int lane = threadIdx.x & 31;
  int row  = blockIdx.x * (blockDim.x >> 5) + wave;
  if (row >= rows) return;
  const float* xr = x + (size_t)row * cols;
  float buf[24];               // cols=768 -> 24 per lane
  float s1 = 0.f, s2 = 0.f;
  int cnt = 0;
  for (int i = lane; i < cols; i += 32) { float v = xr[i]; buf[cnt++] = v; s1 += v; s2 += v * v; }
  for (int off = 16; off >= 1; off >>= 1) {
    s1 += __shfl_xor(s1, off, 32);
    s2 += __shfl_xor(s2, off, 32);
  }
  float mu  = s1 / (float)cols;
  float var = s2 / (float)cols - mu * mu;
  float inv = rsqrtf(var + 1e-5f);
  bf16* orow = out + (size_t)row * cols;
  cnt = 0;
  for (int i = lane; i < cols; i += 32) orow[perm32(i)] = (bf16)((buf[cnt++] - mu) * inv * gamma[i]);
}

// ---------------------------------------------------------------------------
// RoPE + split qkv [B,S,3,H,Dh] fp32 ->
//   q [B,H,S,Dh] bf16 chunk-PERMUTED rows (A-fragment order),
//   k [B,H,S,Dh] bf16 natural rows        (B-fragment of Q.K^T),
//   v [B,H,Dh,S] bf16 TRANSPOSED          (B-fragment of P.V)
// ---------------------------------------------------------------------------
__global__ void k_rope_split(const float* __restrict__ qkv,
                             bf16* __restrict__ qh, bf16* __restrict__ kh, bf16* __restrict__ vT,
                             int Bn, int Hh, int S, int Dm) {
  const int Dh = 64;
  long long tid = (long long)blockIdx.x * blockDim.x + threadIdx.x;
  long long total = (long long)Bn * Hh * S * Dh;
  if (tid >= total) return;
  int d = (int)(tid & 63);
  long long t2 = tid >> 6;
  int s = (int)(t2 % S); t2 /= S;
  int h = (int)(t2 % Hh);
  int b = (int)(t2 / Hh);
  size_t rowbase = ((size_t)b * S + s) * (size_t)(3 * Dm);
  int col = h * Dh + d;
  float q = qkv[rowbase + col];
  float k = qkv[rowbase + Dm + col];
  float v = qkv[rowbase + 2 * Dm + col];
  int dh = d & 31;
  float invf = __powf(10000.f, -((float)(2 * dh)) / 64.f);
  float ang = (float)s * invf;
  float sn, cs;
  __sincosf(ang, &sn, &cs);
  float qr = (d < 32) ? -qkv[rowbase + col + 32]      : qkv[rowbase + col - 32];
  float kr = (d < 32) ? -qkv[rowbase + Dm + col + 32] : qkv[rowbase + Dm + col - 32];
  size_t orow = (((size_t)b * Hh + h) * S + s) * Dh;
  qh[orow + perm32(d)] = (bf16)(q * cs + qr * sn);
  kh[orow + d]         = (bf16)(k * cs + kr * sn);
  vT[((size_t)(b * Hh + h) * Dh + d) * S + s] = (bf16)v;
}

// ---------------------------------------------------------------------------
// bf16 GEMM: C[M,N] = A[M,K] @ BT[N,K]^T (+ res), fp32 out.
// A rows pre-permuted by producers (A-fragment order). 4 waves/WG, 64x64 tile,
// K-step 32. Tiles staged via TDM tensor_load_to_lds, double-buffered.
// ---------------------------------------------------------------------------
__global__ __launch_bounds__(128) void k_gemm_bf16(
    const bf16* __restrict__ A, const bf16* __restrict__ BT,
    const float* __restrict__ res, float* __restrict__ C,
    int Mi, int Ni, int Ki) {
  __shared__ alignas(32) bf16 As[2][64 * 32];
  __shared__ alignas(32) bf16 Bs[2][64 * 32];
  int t = threadIdx.x;
  int lane = t & 31, w = t >> 5;
  int lm = lane & 15;
  int hoff = (lane >= 16) ? 16 : 0;
  int m0 = blockIdx.y * 64, n0 = blockIdx.x * 64;
  const bf16* Abase = A + (size_t)m0 * Ki;
  const bf16* Bbase = BT + (size_t)n0 * Ki;
  unsigned int stride_qw = (unsigned int)(Ki >> 2);
  unsigned int ldsA[2], ldsB[2];
  ldsA[0] = (unsigned int)(uintptr_t)&As[0][0];
  ldsA[1] = (unsigned int)(uintptr_t)&As[1][0];
  ldsB[0] = (unsigned int)(uintptr_t)&Bs[0][0];
  ldsB[1] = (unsigned int)(uintptr_t)&Bs[1][0];
  v8f acc[4] = {{}, {}, {}, {}};

  if (w == 0) {                       // TDM ignores EXEC: one DMA per wave
    tdm_load_tile(Abase, ldsA[0], stride_qw);
    tdm_load_tile(Bbase, ldsB[0], stride_qw);
    __builtin_amdgcn_s_wait_tensorcnt(0);
  }
  __syncthreads();

  int nsteps = Ki >> 5;
  for (int i = 0; i < nsteps; ++i) {
    int cur = i & 1, nxt = cur ^ 1;
    if (w == 0 && i + 1 < nsteps) {   // prefetch next K-tile while computing
      tdm_load_tile(Abase + (i + 1) * 32, ldsA[nxt], stride_qw);
      tdm_load_tile(Bbase + (i + 1) * 32, ldsB[nxt], stride_qw);
    }
    v16bf a = *reinterpret_cast<const v16bf*>(&As[cur][(w * 16 + lm) * 32 + hoff]);
#pragma unroll
    for (int nt = 0; nt < 4; ++nt) {
      v16bf bq = *reinterpret_cast<const v16bf*>(&Bs[cur][(nt * 16 + lm) * 32 + hoff]);
      acc[nt] = __builtin_amdgcn_wmma_f32_16x16x32_bf16(false, a, false, bq,
                                                        (short)0, acc[nt], false, false);
    }
    if (w == 0) __builtin_amdgcn_s_wait_tensorcnt(0);
    __syncthreads();
  }

  bool hi = hoff != 0;
#pragma unroll
  for (int r = 0; r < 8; ++r) {
    int mm = r + (hi ? 8 : 0);
    size_t row = (size_t)m0 + w * 16 + mm;
#pragma unroll
    for (int nt = 0; nt < 4; ++nt) {
      int col = n0 + nt * 16 + lm;
      size_t idx = row * Ni + col;
      C[idx] = acc[nt][r] + (res ? res[idx] : 0.f);
    }
  }
}

// ---------------------------------------------------------------------------
// Sliding-window attention, one wave per (b,h,16-query tile). Window +-64.
// All WMMA fragments are contiguous 32B vector loads.
// ---------------------------------------------------------------------------
__global__ __launch_bounds__(32) void k_attn_swin(
    const bf16* __restrict__ qh, const bf16* __restrict__ kh, const bf16* __restrict__ vT,
    const float* __restrict__ amask, bf16* __restrict__ ctx,
    int Bn, int Hh, int S) {
  const int Dh = 64, W = 64;
  __shared__ alignas(32) float sc[16 * 160];
  __shared__ alignas(32) bf16  pb[16 * 160];   // probs, chunk-permuted per 32-block
  int nq = S >> 4;
  int qt = blockIdx.x % nq;
  int h  = (blockIdx.x / nq) % Hh;
  int b  = blockIdx.x / (nq * Hh);
  int q0 = qt * 16;
  int lane = threadIdx.x & 31;
  bool hi = lane >= 16;
  int lm = lane & 15;
  int hoff = hi ? 16 : 0;
  size_t bh = ((size_t)b * Hh + h) * S;

  for (int i = lane; i < 16 * 160; i += 32) sc[i] = -1e30f;

  // Q fragments: rows already stored in A-fragment order
  const bf16* qrow = qh + (bh + q0 + lm) * Dh;
  v16bf a0 = *reinterpret_cast<const v16bf*>(&qrow[hoff]);
  v16bf a1 = *reinterpret_cast<const v16bf*>(&qrow[32 + hoff]);
  __syncthreads();

  // scores: Q @ K^T over 9 key tiles
  for (int kt = 0; kt < 9; ++kt) {
    int k0 = q0 - W + kt * 16;               // wave-uniform
    if (k0 < 0 || k0 + 16 > S) continue;
    const bf16* krow = kh + (bh + k0 + lm) * Dh;
    v16bf b0 = *reinterpret_cast<const v16bf*>(&krow[hoff]);
    v16bf b1 = *reinterpret_cast<const v16bf*>(&krow[32 + hoff]);
    v8f c = {};
    c = __builtin_amdgcn_wmma_f32_16x16x32_bf16(false, a0, false, b0, (short)0, c, false, false);
    c = __builtin_amdgcn_wmma_f32_16x16x32_bf16(false, a1, false, b1, (short)0, c, false, false);
#pragma unroll
    for (int r = 0; r < 8; ++r) {
      int qm = q0 + r + (hi ? 8 : 0);
      int key = k0 + lm;
      float s = c[r] * 0.125f;               // 1/sqrt(64)
      int dd = qm - key; if (dd < 0) dd = -dd;
      if (dd > W) s = -1e30f;
      else        s += amask[((size_t)b * S + qm) * (size_t)S + key];
      sc[(r + (hi ? 8 : 0)) * 160 + kt * 16 + lm] = s;
    }
  }
  __syncthreads();

  // row softmax over 144 cols; probs written chunk-PERMUTED, zero-padded to 160
  if (lane < 16) {
    int row = lane;
    float mx = -3.0e38f;
    for (int c2 = 0; c2 < 144; ++c2) mx = fmaxf(mx, sc[row * 160 + c2]);
    float sum = 0.f;
    for (int c2 = 0; c2 < 144; ++c2) {
      float e = __expf(sc[row * 160 + c2] - mx);
      sc[row * 160 + c2] = e;
      sum += e;
    }
    float inv = 1.f / sum;
    for (int c2 = 0; c2 < 160; ++c2) {
      float p = (c2 < 144) ? sc[row * 160 + c2] * inv : 0.f;
      pb[row * 160 + perm32(c2)] = (bf16)p;
    }
  }
  __syncthreads();

  // ctx = P @ V : 5 K-blocks of 32 keys x 4 N-tiles of 16 dims (V transposed)
  const bf16* vbase = vT + (size_t)(b * Hh + h) * Dh * S;
  v8f o[4] = {{}, {}, {}, {}};
  const v16bf zv = {};
  for (int blk = 0; blk < 5; ++blk) {
    int key0 = q0 - W + blk * 32;
    v16bf ap = *reinterpret_cast<const v16bf*>(&pb[lm * 160 + blk * 32 + hoff]);
    int kk0 = key0 + hoff;                    // uniform per lane-half, 16-aligned
    bool ok = (kk0 >= 0) && (kk0 + 16 <= S);
#pragma unroll
    for (int nt = 0; nt < 4; ++nt) {
      v16bf bv = ok ? *reinterpret_cast<const v16bf*>(&vbase[(size_t)(nt * 16 + lm) * S + kk0])
                    : zv;
      o[nt] = __builtin_amdgcn_wmma_f32_16x16x32_bf16(false, ap, false, bv,
                                                      (short)0, o[nt], false, false);
    }
  }

  // store ctx as [B,S,H,Dh] bf16, columns in A-fragment order for the Wo GEMM
#pragma unroll
  for (int r = 0; r < 8; ++r) {
    int mm = r + (hi ? 8 : 0);
    size_t orow = (((size_t)b * S + q0 + mm) * Hh + h) * Dh;
    ctx[orow + perm32( 0 + lm)] = (bf16)o[0][r];
    ctx[orow + perm32(16 + lm)] = (bf16)o[1][r];
    ctx[orow + perm32(32 + lm)] = (bf16)o[2][r];
    ctx[orow + perm32(48 + lm)] = (bf16)o[3][r];
  }
}

// ---------------------------------------------------------------------------
// GeGLU: x,gate = split(wi_out); g = gelu_exact(x)*gate -> bf16 (A-frag order)
// ---------------------------------------------------------------------------
__global__ void k_geglu(const float* __restrict__ wiout, bf16* __restrict__ g,
                        int rows, int inter) {
  long long tid = (long long)blockIdx.x * blockDim.x + threadIdx.x;
  if (tid >= (long long)rows * inter) return;
  int row = (int)(tid / inter), col = (int)(tid % inter);
  const float* r = wiout + (size_t)row * (2 * inter);
  float x = r[col], gate = r[inter + col];
  float ge = 0.5f * x * (1.f + erff(x * 0.70710678118654752f));
  g[(size_t)row * inter + perm32(col)] = (bf16)(ge * gate);
}

// ---------------------------------------------------------------------------
extern "C" void kernel_launch(void* const* d_in, const int* in_sizes, int n_in,
                              void* d_out, int out_size, void* d_ws, size_t ws_size,
                              hipStream_t stream) {
  (void)in_sizes; (void)n_in; (void)out_size; (void)ws_size;
  const int Bn = 4, S = 2048, Dm = 768, Hh = 12, I = 1152;
  const int Mrows = Bn * S;  // 8192

  const float* hidden = (const float*)d_in[0];
  const float* amask  = (const float*)d_in[1];
  const float* g1     = (const float*)d_in[2];
  const float* wqkv   = (const float*)d_in[3];
  const float* wo     = (const float*)d_in[4];
  const float* g2     = (const float*)d_in[5];
  const float* wi     = (const float*)d_in[6];
  const float* womlp  = (const float*)d_in[7];
  float* out = (float*)d_out;

  char* base = (char*)d_ws;
  size_t off = 0;
  auto alloc = [&](size_t bytes) -> char* {
    char* p = base + off;
    off = (off + bytes + 255) & ~(size_t)255;
    return p;
  };
  bf16*  wqkv_b  = (bf16*)alloc((size_t)Dm * 3 * Dm * 2);   // [3Dm][Dm] transposed
  bf16*  wo_b    = (bf16*)alloc((size_t)Dm * Dm * 2);       // [Dm][Dm] transposed
  bf16*  wi_b    = (bf16*)alloc((size_t)Dm * 2 * I * 2);    // [2I][Dm] transposed
  bf16*  womlp_b = (bf16*)alloc((size_t)I * Dm * 2);        // [Dm][I] transposed
  bf16*  xln     = (bf16*)alloc((size_t)Mrows * Dm * 2);
  float* bigf    = (float*)alloc((size_t)Mrows * 3 * Dm * 4);  // qkv, later wi-out
  bf16*  qhb     = (bf16*)alloc((size_t)Mrows * Dm * 2);
  bf16*  khb     = (bf16*)alloc((size_t)Mrows * Dm * 2);
  bf16*  vtb     = (bf16*)alloc((size_t)Mrows * Dm * 2);
  bf16*  ctxb    = (bf16*)alloc((size_t)Mrows * Dm * 2);
  float* hbuf    = (float*)alloc((size_t)Mrows * Dm * 4);
  bf16*  nln     = (bf16*)alloc((size_t)Mrows * Dm * 2);
  bf16*  gbuf    = (bf16*)alloc((size_t)Mrows * I * 2);

  auto cdiv = [](long long a, long long b) { return (int)((a + b - 1) / b); };

  // weights -> transposed bf16 [N][K]
  k_cvt_t_bf16<<<cdiv((long long)Dm * 3 * Dm, 256), 256, 0, stream>>>(wqkv,  wqkv_b,  Dm, 3 * Dm);
  k_cvt_t_bf16<<<cdiv((long long)Dm * Dm,     256), 256, 0, stream>>>(wo,    wo_b,    Dm, Dm);
  k_cvt_t_bf16<<<cdiv((long long)Dm * 2 * I,  256), 256, 0, stream>>>(wi,    wi_b,    Dm, 2 * I);
  k_cvt_t_bf16<<<cdiv((long long)I * Dm,      256), 256, 0, stream>>>(womlp, womlp_b, I, Dm);

  // attn pre-norm
  k_layernorm_bf16<<<Mrows / 8, 256, 0, stream>>>(hidden, g1, xln, Mrows, Dm);
  // qkv = xln @ wqkv
  k_gemm_bf16<<<dim3(3 * Dm / 64, Mrows / 64), 128, 0, stream>>>(xln, wqkv_b, nullptr, bigf,
                                                                 Mrows, 3 * Dm, Dm);
  // RoPE + head split (+fragment-order re-layouts)
  k_rope_split<<<cdiv((long long)Bn * Hh * S * 64, 256), 256, 0, stream>>>(bigf, qhb, khb, vtb,
                                                                           Bn, Hh, S, Dm);
  // sliding-window attention
  k_attn_swin<<<Bn * Hh * (S / 16), 32, 0, stream>>>(qhb, khb, vtb, amask, ctxb, Bn, Hh, S);
  // h = hidden + ctx @ wo
  k_gemm_bf16<<<dim3(Dm / 64, Mrows / 64), 128, 0, stream>>>(ctxb, wo_b, hidden, hbuf,
                                                             Mrows, Dm, Dm);
  // mlp pre-norm
  k_layernorm_bf16<<<Mrows / 8, 256, 0, stream>>>(hbuf, g2, nln, Mrows, Dm);
  // wi GEMM
  k_gemm_bf16<<<dim3(2 * I / 64, Mrows / 64), 128, 0, stream>>>(nln, wi_b, nullptr, bigf,
                                                                Mrows, 2 * I, Dm);
  // geglu
  k_geglu<<<cdiv((long long)Mrows * I, 256), 256, 0, stream>>>(bigf, gbuf, Mrows, I);
  // out = h + g @ wo_mlp
  k_gemm_bf16<<<dim3(Dm / 64, Mrows / 64), 128, 0, stream>>>(gbuf, womlp_b, hbuf, out,
                                                             Mrows, Dm, I);
}
